// JambaModel_65687229825534
// MI455X (gfx1250) — matrix operand order repaired and checked by reading, not compiled, not run on hardware
//
#include <hip/hip_runtime.h>
#include <hip/hip_bf16.h>

// ---------------------------------------------------------------------------
// Jamba mini-stack for MI455X (gfx1250).
// All GEMMs run on v_wmma_f32_16x16x32_f16 (f16 operands, f32 accumulate):
// K=32 per WMMA issue vs K=4 for the f32 WMMA -> 8x matrix throughput. Every
// GEMM is NT (B stored [N x K] row-major): weights are transposed once at
// f32->f16 conversion and the V projection writes V^T, so all fragments load
// as contiguous global_load_b128 pairs -- no strided u16 loads, no per-iter
// branches. Each wave owns a 32x32 C tile: 4 WMMAs per 8 b128 loads.
// ---------------------------------------------------------------------------

typedef __attribute__((ext_vector_type(16))) _Float16 v16h;
typedef __attribute__((ext_vector_type(8)))  _Float16 v8h;
typedef __attribute__((ext_vector_type(8)))  float    v8f;

namespace {

constexpr int TQ   = 2048;   // sequence length
constexpr int DM   = 1024;   // model dim
constexpr int DKV  = 256;    // NKV*HD
constexpr int DFF  = 2752;
constexpr int DXZ  = 2048;   // 2*D
constexpr int DXN  = 1056;   // D + 2N
constexpr int HD   = 64;
constexpr int NH   = 16;
constexpr size_t TD = (size_t)TQ * DM;

// ---- f16 transposed-weight arena element offsets ---------------------------
constexpr size_t WQ_O   = 0;                               // [D x D]
constexpr size_t WK_O   = WQ_O   + (size_t)DM * DM;        // [DKV x D]
constexpr size_t WV_O   = WK_O   + (size_t)DM * DKV;
constexpr size_t WO_O   = WV_O   + (size_t)DM * DKV;
constexpr size_t WG_O   = WO_O   + (size_t)DM * DM;        // [DFF x D]
constexpr size_t WU_O   = WG_O   + (size_t)DM * DFF;
constexpr size_t WD_O   = WU_O   + (size_t)DM * DFF;       // [D x DFF]
constexpr size_t WIN_O  = WD_O   + (size_t)DFF * DM;       // 3 x [DXZ x D]
constexpr size_t WX_O   = WIN_O  + 3 * (size_t)DM * DXZ;   // 3 x [DXN x D]
constexpr size_t WOUT_O = WX_O   + 3 * (size_t)DM * DXN;   // 3 x [D x D]
constexpr size_t WH_ELEMS = WOUT_O + 3 * (size_t)DM * DM;  // 23,756,800

// ---- workspace byte layout --------------------------------------------------
constexpr size_t XBUF_B  = 0;                               // residual stream, f32, 8MB
constexpr size_t WH_B    = XBUF_B + TD * 4;                 // f16 weights ~47.5MB
constexpr size_t P_B     = (WH_B + WH_ELEMS * 2 + 255) & ~(size_t)255;
constexpr size_t XNH_B   = P_B;                             // rmsnorm out f16 (4MB)
constexpr size_t SLOT1_B = XNH_B   + 4194304;  // attn: q_h | ffn: g f32 | mamba: xz f32
constexpr size_t SLOT2_B = SLOT1_B + 23068672; // attn: k_h,vt_h | ffn: u f32 | mamba: xcF,xcH
constexpr size_t SLOT3_B = SLOT2_B + 23068672; // attn: s f32 | ffn: gu_h | mamba: proj
constexpr size_t SLOT4_B = SLOT3_B + 16777216; // attn: p f16 | mamba: ys f32
constexpr size_t SLOT5_B = SLOT4_B + 8388608;  // attn: o_h  | mamba: ym_h
} // namespace

__device__ __forceinline__ float siluf(float v) { return v / (1.0f + __expf(-v)); }

// A fragment (16x32): lane l16 = row, elems 0-7 = K[sel*8..+7], 8-15 = K[16+sel*8..+7]
__device__ __forceinline__ v16h load_frag_a(const _Float16* p) {
  const v8h x = *(const v8h*)(p);
  const v8h y = *(const v8h*)(p + 16);
  v16h r;
#pragma unroll
  for (int i = 0; i < 8; ++i) { r[i] = x[i]; r[i + 8] = y[i]; }
  return r;
}
// B fragment (32x16) from B^T [N x K]: lane l16 = col n, elems 0-7 = K[sel*8..+7], 8-15 = K[16+sel*8..+7]
__device__ __forceinline__ v16h load_frag_b(const _Float16* p) {
  const v8h x = *(const v8h*)(p);
  const v8h y = *(const v8h*)(p + 16);
  v16h r;
#pragma unroll
  for (int i = 0; i < 8; ++i) { r[i] = x[i]; r[i + 8] = y[i]; }
  return r;
}

// ---------------------------------------------------------------------------
// NT WMMA GEMM: C[M,N] = A[M,K] . Bt[N,K]^T (+ residual).
// 128 threads = 4 waves stacked in M; each wave computes a 32x32 C tile
// (2 A frags x 2 B frags -> 4 WMMAs per K-step). Block tile = 128(M) x 32(N).
// wHmode: 0 = none, 1 = f16 row-major, 2 = f16 transposed (col-major store).
// ---------------------------------------------------------------------------
__global__ void __launch_bounds__(128)
gemm_nt_wmma(const _Float16* __restrict__ A, int lda,
             const _Float16* __restrict__ Bt, int ldb,
             float* __restrict__ outF, int ldc, int writeF,
             _Float16* __restrict__ outH, int ldch, int wHmode,
             const float* __restrict__ resF, int ldr, int addRes,
             int M, int N, int K)
{
  const int wave = threadIdx.x >> 5;
  const int lane = threadIdx.x & 31;
  const int l16  = lane & 15;
  const int sel  = lane >> 4;
  const int mTile = blockIdx.y * 128 + wave * 32;
  const int nTile = blockIdx.x * 32;
  if (mTile >= M || nTile >= N) return;

  const _Float16* a0p = A  + (size_t)(mTile + l16) * lda + sel * 8;
  const _Float16* a1p = a0p + (size_t)16 * lda;
  const _Float16* b0p = Bt + (size_t)(nTile + l16) * ldb + sel * 8;
  const _Float16* b1p = b0p + (size_t)16 * ldb;

  v8f c00 = {}, c01 = {}, c10 = {}, c11 = {};
  for (int k0 = 0; k0 < K; k0 += 32) {
    const v16h a0 = load_frag_a(a0p + k0);
    const v16h a1 = load_frag_a(a1p + k0);
    const v16h b0 = load_frag_b(b0p + k0);
    const v16h b1 = load_frag_b(b1p + k0);
    c00 = __builtin_amdgcn_wmma_f32_16x16x32_f16(false, a0, false, b0, (short)0, c00, false, false);
    c01 = __builtin_amdgcn_wmma_f32_16x16x32_f16(false, a0, false, b1, (short)0, c01, false, false);
    c10 = __builtin_amdgcn_wmma_f32_16x16x32_f16(false, a1, false, b0, (short)0, c10, false, false);
    c11 = __builtin_amdgcn_wmma_f32_16x16x32_f16(false, a1, false, b1, (short)0, c11, false, false);
  }

  const v8f* acc[4] = { &c00, &c01, &c10, &c11 };
#pragma unroll
  for (int t = 0; t < 4; ++t) {
    const int dm = (t >> 1) * 16, dn = (t & 1) * 16;
#pragma unroll
    for (int v = 0; v < 8; ++v) {
      const int row = mTile + dm + sel * 8 + v;
      const int col = nTile + dn + l16;
      float val = (*acc[t])[v];
      if (addRes) val += resF[(size_t)row * ldr + col];
      if (writeF) outF[(size_t)row * ldc + col] = val;
      if (wHmode == 1) outH[(size_t)row * ldch + col] = (_Float16)val;
      else if (wHmode == 2) outH[(size_t)col * ldch + row] = (_Float16)val;
    }
  }
}

// ---------------------------------------------------------------------------
__global__ void __launch_bounds__(256)
rmsnorm_f16(const float* __restrict__ x, const float* __restrict__ w,
            _Float16* __restrict__ y, int D)
{
  __shared__ float red[256];
  const int row = blockIdx.x;
  const int tid = threadIdx.x;
  const float* xr = x + (size_t)row * D;
  float s = 0.f;
  for (int i = tid; i < D; i += 256) { float v = xr[i]; s += v * v; }
  red[tid] = s; __syncthreads();
  for (int off = 128; off > 0; off >>= 1) {
    if (tid < off) red[tid] += red[tid + off];
    __syncthreads();
  }
  const float scale = rsqrtf(red[0] / (float)D + 1e-6f);
  _Float16* yr = y + (size_t)row * D;
  for (int i = tid; i < D; i += 256) yr[i] = (_Float16)(xr[i] * scale * w[i]);
}

// Causal softmax over one score row; writes f16 probabilities (zeros past diag).
__global__ void __launch_bounds__(256)
softmax_causal(const float* __restrict__ s, _Float16* __restrict__ p,
               int T, float scale)
{
  __shared__ float red[256];
  const int row = blockIdx.x;
  const int tid = threadIdx.x;
  const int len = row + 1;
  const float* sr = s + (size_t)row * T;
  _Float16* pr = p + (size_t)row * T;

  float m = -3.4e38f;
  for (int j = tid; j < len; j += 256) m = fmaxf(m, sr[j] * scale);
  red[tid] = m; __syncthreads();
  for (int off = 128; off > 0; off >>= 1) {
    if (tid < off) red[tid] = fmaxf(red[tid], red[tid + off]);
    __syncthreads();
  }
  m = red[0]; __syncthreads();

  float sum = 0.f;
  for (int j = tid; j < len; j += 256) sum += __expf(sr[j] * scale - m);
  red[tid] = sum; __syncthreads();
  for (int off = 128; off > 0; off >>= 1) {
    if (tid < off) red[tid] += red[tid + off];
    __syncthreads();
  }
  const float inv = 1.0f / red[0];
  for (int j = tid; j < len; j += 256)
    pr[j] = (_Float16)(__expf(sr[j] * scale - m) * inv);
  for (int j = len + tid; j < T; j += 256) pr[j] = (_Float16)0.f;
}

// Depthwise causal conv (K=4) + bias + SiLU. Input xb = xz[:, :D], stride ldx.
__global__ void __launch_bounds__(256)
dwconv_silu(const float* __restrict__ xz, int ldx,
            const float* __restrict__ w, const float* __restrict__ b,
            float* __restrict__ xcF, _Float16* __restrict__ xcH, int T, int D)
{
  const int idx = blockIdx.x * 256 + threadIdx.x;
  if (idx >= T * D) return;
  const int t = idx / D, d = idx - t * D;
  float acc = b[d];
#pragma unroll
  for (int k = 0; k < 4; ++k) {
    const int tt = t + k - 3;
    if (tt >= 0) acc += xz[(size_t)tt * ldx + d] * w[d * 4 + k];
  }
  const float y = siluf(acc);
  xcF[idx] = y;
  xcH[idx] = (_Float16)y;
}

__global__ void __launch_bounds__(256)
ffn_silumul(const float* __restrict__ g, const float* __restrict__ u,
            _Float16* __restrict__ o, int n)
{
  const int i = blockIdx.x * 256 + threadIdx.x;
  if (i >= n) return;
  o[i] = (_Float16)(siluf(g[i]) * u[i]);
}

// ym = ys * silu(z), z = xz[:, D:2D]
__global__ void __launch_bounds__(256)
mamba_gate(const float* __restrict__ ys, const float* __restrict__ xz,
           _Float16* __restrict__ o, int n)
{
  const int i = blockIdx.x * 256 + threadIdx.x;
  if (i >= n) return;
  const int t = i >> 10, c = i & 1023;
  const float z = xz[(size_t)t * DXZ + DM + c];
  o[i] = (_Float16)(ys[i] * siluf(z));
}

// Sequential selective-scan: 1024 channels over 32 single-wave blocks,
// state h[16] in registers, B_t/C_t staged through LDS each step.
__global__ void __launch_bounds__(32)
ssm_scan(const float* __restrict__ proj, const float* __restrict__ xc,
         const float* __restrict__ a_log, const float* __restrict__ dskip,
         float* __restrict__ ys, float* __restrict__ hout, int T)
{
  __shared__ float Bs[16], Cs[16];
  const int tid = threadIdx.x;
  const int d = blockIdx.x * 32 + tid;
  float A[16], h[16];
#pragma unroll
  for (int n = 0; n < 16; ++n) { A[n] = -__expf(a_log[d * 16 + n]); h[n] = 0.f; }
  const float dsk = dskip[d];
  for (int t = 0; t < T; ++t) {
    const float* pr = proj + (size_t)t * DXN;
    if (tid < 16) Bs[tid] = pr[DM + tid];
    else          Cs[tid - 16] = pr[DM + 16 + (tid - 16)];
    __syncthreads();
    const float pd = pr[d];
    const float delta = pd > 20.f ? pd : log1pf(__expf(pd));
    const float xv = xc[(size_t)t * DM + d];
    float acc = 0.f;
#pragma unroll
    for (int n = 0; n < 16; ++n) {
      h[n] = __expf(delta * A[n]) * h[n] + (delta * Bs[n]) * xv;
      acc += h[n] * Cs[n];
    }
    ys[(size_t)t * DM + d] = acc + dsk * xv;
    __syncthreads();
  }
#pragma unroll
  for (int n = 0; n < 16; ++n) hout[d * 16 + n] = h[n];
}

// Tiled transpose + f32->f16: src [R x C] f32 row-major -> dst [C x R] f16.
__global__ void __launch_bounds__(256)
cvt_t_f32_f16(const float* __restrict__ src, _Float16* __restrict__ dst,
              int R, int C)
{
  __shared__ float tile[32][33];
  const int tx = threadIdx.x, ty = threadIdx.y;
  const int c0 = blockIdx.x * 32, r0 = blockIdx.y * 32;
#pragma unroll
  for (int j = 0; j < 32; j += 8)
    tile[ty + j][tx] = src[(size_t)(r0 + ty + j) * C + (c0 + tx)];
  __syncthreads();
#pragma unroll
  for (int j = 0; j < 32; j += 8)
    dst[(size_t)(c0 + ty + j) * R + (r0 + tx)] = (_Float16)tile[tx][ty + j];
}

// ---------------------------------------------------------------------------
extern "C" void kernel_launch(void* const* d_in, const int* in_sizes, int n_in,
                              void* d_out, int out_size, void* d_ws, size_t ws_size,
                              hipStream_t stream)
{
  (void)in_sizes; (void)n_in; (void)out_size; (void)ws_size;

  const float* x_in    = (const float*)d_in[0];
  const float* attn_nw = (const float*)d_in[1];
  const float* ffn_nw  = (const float*)d_in[2];
  const float* Wq      = (const float*)d_in[3];
  const float* Wk      = (const float*)d_in[4];
  const float* Wv      = (const float*)d_in[5];
  const float* Wo      = (const float*)d_in[6];
  const float* Wg      = (const float*)d_in[7];
  const float* Wu      = (const float*)d_in[8];
  const float* Wd      = (const float*)d_in[9];
  const float* m_nw    = (const float*)d_in[10];
  const float* Win     = (const float*)d_in[11];
  const float* conv_w  = (const float*)d_in[12];
  const float* conv_b  = (const float*)d_in[13];
  const float* Wx      = (const float*)d_in[14];
  const float* A_log   = (const float*)d_in[15];
  const float* Dp      = (const float*)d_in[16];
  const float* Wout    = (const float*)d_in[17];

  char* ws = (char*)d_ws;
  float*    xbuf = (float*)(ws + XBUF_B);
  _Float16* wh   = (_Float16*)(ws + WH_B);
  _Float16* xn_h = (_Float16*)(ws + XNH_B);

  // transpose-convert: src [R x C] f32 -> wh+woff [C x R] f16
  auto cvtT = [&](const float* src, size_t woff, int R, int C) {
    dim3 g((unsigned)(C / 32), (unsigned)(R / 32));
    cvt_t_f32_f16<<<g, dim3(32, 8), 0, stream>>>(src, wh + woff, R, C);
  };
  auto gemm = [&](const _Float16* A, int lda, const _Float16* Bt, int ldb,
                  float* oF, int ldc, int wF, _Float16* oH, int ldch, int wHm,
                  const float* rF, int ldr, int aR, int M, int N, int K) {
    dim3 g((unsigned)(N / 32), (unsigned)(M / 128));
    gemm_nt_wmma<<<g, 128, 0, stream>>>(A, lda, Bt, ldb, oF, ldc, wF,
                                        oH, ldch, wHm, rF, ldr, aR, M, N, K);
  };

  // ---- convert all weights to f16, transposed, once ----
  cvtT(Wq, WQ_O, DM, DM);
  cvtT(Wk, WK_O, DM, DKV);
  cvtT(Wv, WV_O, DM, DKV);
  cvtT(Wo, WO_O, DM, DM);
  cvtT(Wg, WG_O, DM, DFF);
  cvtT(Wu, WU_O, DM, DFF);
  cvtT(Wd, WD_O, DFF, DM);
  for (int mi = 0; mi < 3; ++mi) {
    cvtT(Win  + (size_t)mi * DM * DXZ, WIN_O  + (size_t)mi * DM * DXZ, DM, DXZ);
    cvtT(Wx   + (size_t)mi * DM * DXN, WX_O   + (size_t)mi * DM * DXN, DM, DXN);
    cvtT(Wout + (size_t)mi * DM * DM,  WOUT_O + (size_t)mi * DM * DM,  DM, DM);
  }

  // residual stream
  hipMemcpyAsync(xbuf, x_in, TD * sizeof(float), hipMemcpyDeviceToDevice, stream);

  const unsigned tdBlocks = (unsigned)((TD + 255) / 256);

  // =========================== layer 0: attention ===========================
  {
    _Float16* q_h  = (_Float16*)(ws + SLOT1_B);                          // [T x D]
    _Float16* k_h  = (_Float16*)(ws + SLOT2_B);                          // [T x DKV]
    _Float16* vt_h = (_Float16*)(ws + SLOT2_B + (size_t)TQ * DKV * 2);   // [DKV x T]
    float*    sbuf = (float*)(ws + SLOT3_B);      // 2048x2048 scores (one head)
    _Float16* pbuf = (_Float16*)(ws + SLOT4_B);   // 2048x2048 probs f16
    _Float16* o_h  = (_Float16*)(ws + SLOT5_B);   // [T x D] f16

    rmsnorm_f16<<<TQ, 256, 0, stream>>>(xbuf, attn_nw, xn_h, DM);
    gemm(xn_h, DM, wh + WQ_O, DM, nullptr, 0, 0, q_h, DM,  1, nullptr, 0, 0, TQ, DM,  DM);
    gemm(xn_h, DM, wh + WK_O, DM, nullptr, 0, 0, k_h, DKV, 1, nullptr, 0, 0, TQ, DKV, DM);
    // V projection stored transposed: vt_h[d][t], so P.V is NT as well.
    gemm(xn_h, DM, wh + WV_O, DM, nullptr, 0, 0, vt_h, TQ, 2, nullptr, 0, 0, TQ, DKV, DM);

    for (int hh = 0; hh < NH; ++hh) {
      const int kv = hh >> 2;  // NREP = 4
      // S = Q_h . K_kv^T   (Bt = k_h rows, K = HD = 64)
      gemm(q_h + hh * HD, DM, k_h + kv * HD, DKV,
           sbuf, TQ, 1, nullptr, 0, 0, nullptr, 0, 0, TQ, TQ, HD);
      softmax_causal<<<TQ, 256, 0, stream>>>(sbuf, pbuf, TQ, 0.125f /* HD^-0.5 */);
      // O_h = P . V_kv     (Bt = V^T [HD x T], K = T)
      gemm(pbuf, TQ, vt_h + (size_t)kv * HD * TQ, TQ,
           nullptr, 0, 0, o_h + hh * HD, DM, 1, nullptr, 0, 0, TQ, HD, TQ);
    }
    // x += O . Wo
    gemm(o_h, DM, wh + WO_O, DM, xbuf, DM, 1, nullptr, 0, 0, xbuf, DM, 1, TQ, DM, DM);
  }

  // ================================ FFN =====================================
  {
    float*    gbuf = (float*)(ws + SLOT1_B);
    float*    ubuf = (float*)(ws + SLOT2_B);
    _Float16* gu_h = (_Float16*)(ws + SLOT3_B);
    const int nGU = TQ * DFF;

    rmsnorm_f16<<<TQ, 256, 0, stream>>>(xbuf, ffn_nw, xn_h, DM);
    gemm(xn_h, DM, wh + WG_O, DM, gbuf, DFF, 1, nullptr, 0, 0, nullptr, 0, 0, TQ, DFF, DM);
    gemm(xn_h, DM, wh + WU_O, DM, ubuf, DFF, 1, nullptr, 0, 0, nullptr, 0, 0, TQ, DFF, DM);
    ffn_silumul<<<(unsigned)((nGU + 255) / 256), 256, 0, stream>>>(gbuf, ubuf, gu_h, nGU);
    // x += (silu(g)*u) . Wd
    gemm(gu_h, DFF, wh + WD_O, DFF, xbuf, DM, 1, nullptr, 0, 0, xbuf, DM, 1, TQ, DM, DFF);
  }

  // ============================ Mamba layers ================================
  float* hs_out = (float*)d_out + TD;  // (3,1,1024,16) after final x
  for (int mi = 0; mi < 3; ++mi) {
    float*    xz   = (float*)(ws + SLOT1_B);                 // 2048x2048
    float*    xcF  = (float*)(ws + SLOT2_B);                 // 2048x1024
    _Float16* xcH  = (_Float16*)(ws + SLOT2_B + TD * 4);
    float*    proj = (float*)(ws + SLOT3_B);                 // 2048x1056
    float*    ysb  = (float*)(ws + SLOT4_B);                 // 2048x1024
    _Float16* ym_h = (_Float16*)(ws + SLOT5_B);

    rmsnorm_f16<<<TQ, 256, 0, stream>>>(xbuf, m_nw + (size_t)mi * DM, xn_h, DM);
    gemm(xn_h, DM, wh + WIN_O + (size_t)mi * DM * DXZ, DM,
         xz, DXZ, 1, nullptr, 0, 0, nullptr, 0, 0, TQ, DXZ, DM);
    dwconv_silu<<<tdBlocks, 256, 0, stream>>>(xz, DXZ,
        conv_w + (size_t)mi * DM * 4, conv_b + (size_t)mi * DM, xcF, xcH, TQ, DM);
    gemm(xcH, DM, wh + WX_O + (size_t)mi * DM * DXN, DM,
         proj, DXN, 1, nullptr, 0, 0, nullptr, 0, 0, TQ, DXN, DM);
    ssm_scan<<<32, 32, 0, stream>>>(proj, xcF,
        A_log + (size_t)mi * DM * 16, Dp + (size_t)mi * DM,
        ysb, hs_out + (size_t)mi * DM * 16, TQ);
    mamba_gate<<<tdBlocks, 256, 0, stream>>>(ysb, xz, ym_h, (int)TD);
    // x += (ys * silu(z)) . Wout
    gemm(ym_h, DM, wh + WOUT_O + (size_t)mi * DM * DM, DM,
         xbuf, DM, 1, nullptr, 0, 0, xbuf, DM, 1, TQ, DM, DM);
  }

  hipMemcpyAsync(d_out, xbuf, TD * sizeof(float), hipMemcpyDeviceToDevice, stream);
}
